// Spatial2DNATBlock_46557445489234
// MI455X (gfx1250) — compile-verified, matrix operand order checked by reading, not run on hardware
//
#include <hip/hip_runtime.h>
#include <hip/hip_bf16.h>
#include <cstdint>

typedef __bf16 bf16;
typedef __attribute__((ext_vector_type(16))) __bf16 v16bf;
typedef __attribute__((ext_vector_type(8)))  __bf16 v8bf;
typedef __attribute__((ext_vector_type(8)))  float  v8f;

#define BB    2
#define CCH   128
#define HH    128
#define WW    128
#define HWSZ  (HH*WW)
#define MTOK  (BB*HWSZ)
#define NHD   8
#define HC    64
#define WC    64
#define HIDR  170
#define HIDP  192

__device__ inline float wred_sum(float v){
#pragma unroll
  for (int m = 16; m > 0; m >>= 1) v += __shfl_xor(v, m, 32);
  return v;
}
__device__ inline float wred_max(float v){
#pragma unroll
  for (int m = 16; m > 0; m >>= 1) v = fmaxf(v, __shfl_xor(v, m, 32));
  return v;
}

// gfx1250 async global->LDS DMA (per-lane 16B), tracked by ASYNCcnt.
// vdst = wave-relative LDS byte address (low 32 bits of the flat pointer),
// vaddr = 64-bit global address.
__device__ inline void async_ld128(unsigned lds_off, const void* gptr){
  asm volatile("global_load_async_to_lds_b128 %0, %1, off"
               :: "v"(lds_off), "v"((unsigned long long)(uintptr_t)gptr)
               : "memory");
}
__device__ inline void wait_async0(){
  asm volatile("s_wait_asynccnt 0x0" ::: "memory");
}

// ---------------- transpose + LN1 -> xh (f32 residual), xn (bf16) ----------------
__global__ __launch_bounds__(256) void k_ln1(const float* __restrict__ x,
                                             const float* __restrict__ g,
                                             const float* __restrict__ bta,
                                             float* __restrict__ xh,
                                             bf16* __restrict__ xn){
  __shared__ float tile[128][65];
  int tid = threadIdx.x;
  int t0  = blockIdx.x * 64;
  int b   = t0 / HWSZ;
  int hw0 = t0 % HWSZ;
  const float* xb = x + (size_t)b * CCH * HWSZ + hw0;
  for (int e = tid; e < 128*64; e += 256){
    int c = e >> 6, t = e & 63;
    tile[c][t] = xb[(size_t)c * HWSZ + t];
  }
  __syncthreads();
  int wv = tid >> 5, lane = tid & 31;
  for (int tt = wv*8; tt < wv*8 + 8; ++tt){
    float v[4]; float s = 0.f;
#pragma unroll
    for (int j = 0; j < 4; ++j){ v[j] = tile[lane + 32*j][tt]; s += v[j]; }
    float mu = wred_sum(s) * (1.f/128.f);
    float q = 0.f;
#pragma unroll
    for (int j = 0; j < 4; ++j){ float d = v[j]-mu; q += d*d; }
    float r = rsqrtf(wred_sum(q) * (1.f/128.f) + 1e-6f);
    size_t base = (size_t)(t0 + tt) * 128;
#pragma unroll
    for (int j = 0; j < 4; ++j){
      int c = lane + 32*j;
      xh[base + c] = v[j];
      xn[base + c] = (bf16)(((v[j]-mu)*r) * g[c] + bta[c]);
    }
  }
}

// ---------------- bf16 WMMA GEMM: C[M,N] = A[M,K] * Bw[N,K]^T + bias --------------
// block: 256 thr (8 waves), tile 256x64, BK=32.
// wave -> 32 rows x 64 cols = 2x4 wmma tiles (8 WMMAs per k-step).
// Tiles staged with gfx1250 async global->LDS DMA.
__global__ __launch_bounds__(256) void k_gemm(const bf16* __restrict__ A,
                                              const bf16* __restrict__ Bw,
                                              const float* __restrict__ bias,
                                              float* __restrict__ Co,
                                              int K, int N){
  __shared__ alignas(16) bf16 sA[256*32];   // 16 KB
  __shared__ alignas(16) bf16 sB[64*32];    //  4 KB
  int tid  = threadIdx.x;
  int row0 = blockIdx.x * 256;
  int col0 = blockIdx.y * 64;
  int wv = tid >> 5, lane = tid & 31;
  int wr = wv * 32;
  int r  = lane & 15, kh = lane >> 4;

  v8f acc[2][4];
#pragma unroll
  for (int i = 0; i < 2; ++i)
#pragma unroll
    for (int j = 0; j < 4; ++j)
#pragma unroll
      for (int e = 0; e < 8; ++e) acc[i][j][e] = 0.f;

  for (int k0 = 0; k0 < K; k0 += 32){
    __syncthreads();   // prior iteration's LDS reads retired across all waves
    // A tile: 256x32 bf16 = 1024 x 16B chunks -> 4 async DMAs per thread
#pragma unroll
    for (int p = 0; p < 4; ++p){
      int e  = p*256 + tid;
      int rr = e >> 2, kc = e & 3;
      async_ld128((unsigned)(uintptr_t)&sA[rr*32 + kc*8],
                  A + (size_t)(row0+rr)*K + k0 + kc*8);
    }
    // B tile: 64x32 bf16 = 256 x 16B chunks -> 1 async DMA per thread
    {
      int rr = tid >> 2, kc = tid & 3;
      async_ld128((unsigned)(uintptr_t)&sB[rr*32 + kc*8],
                  Bw + (size_t)(col0+rr)*K + k0 + kc*8);
    }
    if (k0 + 32 < K){
      int rr = tid >> 2;
      __builtin_prefetch(A + (size_t)(row0+rr)*K + k0 + 32, 0, 0);
    }
    wait_async0();     // our DMAs landed in LDS
    __syncthreads();   // everyone's DMAs visible

    v16bf af[2], bfr[4];
#pragma unroll
    for (int i = 0; i < 2; ++i){
      int ar = wr + i*16 + r;
      v8bf alo = *(const v8bf*)(&sA[ar*32 + kh*8]);
      v8bf ahi = *(const v8bf*)(&sA[ar*32 + 16 + kh*8]);
      af[i] = __builtin_shufflevector(alo, ahi, 0,1,2,3,4,5,6,7,8,9,10,11,12,13,14,15);
    }
#pragma unroll
    for (int j = 0; j < 4; ++j){
      int bc = j*16 + r;
      v8bf blo = *(const v8bf*)(&sB[bc*32 + kh*8]);
      v8bf bhi = *(const v8bf*)(&sB[bc*32 + 16 + kh*8]);
      bfr[j] = __builtin_shufflevector(blo, bhi, 0,1,2,3,4,5,6,7,8,9,10,11,12,13,14,15);
    }
#pragma unroll
    for (int i = 0; i < 2; ++i)
#pragma unroll
      for (int j = 0; j < 4; ++j)
        acc[i][j] = __builtin_amdgcn_wmma_f32_16x16x32_bf16(
            false, af[i], false, bfr[j], (short)0, acc[i][j], false, false);
  }

  int mhalf = (lane >> 4) * 8;
  int nl    = lane & 15;
#pragma unroll
  for (int i = 0; i < 2; ++i)
#pragma unroll
    for (int j = 0; j < 4; ++j){
      int nn = col0 + j*16 + nl;
      float bv = bias ? bias[nn] : 0.f;
#pragma unroll
      for (int e = 0; e < 8; ++e){
        int mm = row0 + wr + i*16 + mhalf + e;
        Co[(size_t)mm * N + nn] = acc[i][j][e] + bv;
      }
    }
}

// ---------------- LN over 128 channels of one row -> bf16 -------------------------
__device__ inline void ln128_to_bf16(const float* __restrict__ row,
                                     const float* __restrict__ g,
                                     const float* __restrict__ bta,
                                     bf16* __restrict__ dst, int lane){
  float v[4]; float s = 0.f;
#pragma unroll
  for (int j = 0; j < 4; ++j){ v[j] = row[lane + 32*j]; s += v[j]; }
  float mu = wred_sum(s) * (1.f/128.f);
  float q = 0.f;
#pragma unroll
  for (int j = 0; j < 4; ++j){ float d = v[j]-mu; q += d*d; }
  float r = rsqrtf(wred_sum(q) * (1.f/128.f) + 1e-6f);
#pragma unroll
  for (int j = 0; j < 4; ++j){
    int c = lane + 32*j;
    dst[c] = (bf16)(((v[j]-mu)*r) * g[c] + bta[c]);
  }
}

// ---------------- q/k LN + stride-2 subsample of k,v ------------------------------
__global__ __launch_bounds__(256) void k_qkprep(const float* __restrict__ qkv,
        const float* __restrict__ qg, const float* __restrict__ qb_,
        const float* __restrict__ kg, const float* __restrict__ kb_,
        bf16* __restrict__ qo, bf16* __restrict__ ko, bf16* __restrict__ vo){
  int tid = threadIdx.x; int wv = tid >> 5, lane = tid & 31;
  size_t t = (size_t)blockIdx.x * 8 + wv;
  const float* row = qkv + t * 384;
  ln128_to_bf16(row, qg, qb_, qo + t*128, lane);
  int b  = (int)(t / HWSZ);
  int hw = (int)(t % HWSZ);
  int hy = hw / WW, wx = hw % WW;
  if (((hy | wx) & 1) == 0){   // wave-uniform
    size_t ci = ((size_t)b*HC*WC + (size_t)(hy>>1)*WC + (wx>>1)) * 128;
    ln128_to_bf16(row + 128, kg, kb_, ko + ci, lane);
#pragma unroll
    for (int j = 0; j < 4; ++j){ int c = lane + 32*j; vo[ci + c] = (bf16)row[256 + c]; }
  }
}

// ---------------- neighborhood attention: one block per coarse cell ---------------
__global__ __launch_bounds__(256) void k_attn(const bf16* __restrict__ qm,
                                              const bf16* __restrict__ km,
                                              const bf16* __restrict__ vm,
                                              bf16* __restrict__ ao){
  __shared__ alignas(16) bf16 sk[64][128];
  __shared__ alignas(16) bf16 sv[64][128];
  __shared__ alignas(16) bf16 sq[4][128];
  __shared__ float sp[8][4][64];
  __shared__ int   smask[64];
  int tid = threadIdx.x;
  int blk = blockIdx.x;
  int b   = blk / (HC*WC);
  int rem = blk % (HC*WC);
  int ic  = rem / WC, jc = rem % WC;

  for (int e = tid; e < 4*128; e += 256){
    int qi = e >> 7, c = e & 127;
    int hy = 2*ic + (qi >> 1), wx = 2*jc + (qi & 1);
    sq[qi][c] = qm[((size_t)b*HWSZ + (size_t)hy*WW + wx)*128 + c];
  }
  {
    int n = tid >> 2, part = tid & 3;
    int di = n >> 3, dj = n & 7;
    int yy = ic + 2*(di - 4), xx = jc + 2*(dj - 4);
    bool valid = (yy >= 0 && yy < HC && xx >= 0 && xx < WC);
    if (part == 0) smask[n] = valid ? 1 : 0;
    int c0 = part * 32;
    uint4* kd = (uint4*)(&sk[n][c0]);
    uint4* vd = (uint4*)(&sv[n][c0]);
    if (valid){
      size_t base = ((size_t)b*HC*WC + (size_t)yy*WC + xx) * 128 + c0;
      const uint4* ks = (const uint4*)(km + base);
      const uint4* vs = (const uint4*)(vm + base);
#pragma unroll
      for (int u = 0; u < 4; ++u){ kd[u] = ks[u]; vd[u] = vs[u]; }
    } else {
      uint4 z = make_uint4(0u,0u,0u,0u);
#pragma unroll
      for (int u = 0; u < 4; ++u){ kd[u] = z; vd[u] = z; }
    }
  }
  __syncthreads();

  int h = tid >> 5, lane = tid & 31;
  float sc[2][4];
#pragma unroll
  for (int rep = 0; rep < 2; ++rep){
    int nn = rep*32 + lane;
    float d0=0.f, d1=0.f, d2=0.f, d3=0.f;
#pragma unroll
    for (int c = 0; c < 16; ++c){
      float kv = (float)sk[nn][h*16 + c];
      d0 += (float)sq[0][h*16 + c] * kv;
      d1 += (float)sq[1][h*16 + c] * kv;
      d2 += (float)sq[2][h*16 + c] * kv;
      d3 += (float)sq[3][h*16 + c] * kv;
    }
    bool m = smask[nn] != 0;
    const float ninf = -1e30f;
    sc[rep][0] = m ? d0*0.25f : ninf;   // dh^-0.5 = 1/sqrt(16)
    sc[rep][1] = m ? d1*0.25f : ninf;
    sc[rep][2] = m ? d2*0.25f : ninf;
    sc[rep][3] = m ? d3*0.25f : ninf;
  }
#pragma unroll
  for (int q = 0; q < 4; ++q){
    float mx  = wred_max(fmaxf(sc[0][q], sc[1][q]));
    float e0  = __expf(sc[0][q] - mx);
    float e1  = __expf(sc[1][q] - mx);
    float inv = 1.f / wred_sum(e0 + e1);
    sp[h][q][lane]      = e0 * inv;
    sp[h][q][32 + lane] = e1 * inv;
  }
  __syncthreads();

  int qp = lane >> 4, dd = lane & 15;
  float a0 = 0.f, a1 = 0.f;
  for (int nn = 0; nn < 64; ++nn){
    float vv = (float)sv[nn][h*16 + dd];
    a0 += sp[h][qp*2    ][nn] * vv;
    a1 += sp[h][qp*2 + 1][nn] * vv;
  }
  {
    int q0 = qp*2, q1 = qp*2 + 1;
    int hy0 = 2*ic + (q0 >> 1), wx0 = 2*jc + (q0 & 1);
    int hy1 = 2*ic + (q1 >> 1), wx1 = 2*jc + (q1 & 1);
    ao[((size_t)b*HWSZ + (size_t)hy0*WW + wx0)*128 + h*16 + dd] = (bf16)a0;
    ao[((size_t)b*HWSZ + (size_t)hy1*WW + wx1)*128 + h*16 + dd] = (bf16)a1;
  }
}

// ---------------- residual 1 + LN2 ------------------------------------------------
__global__ __launch_bounds__(256) void k_resid1(const float* __restrict__ pout,
        const float* __restrict__ ls1, const float* __restrict__ g,
        const float* __restrict__ bta, float* __restrict__ xh,
        bf16* __restrict__ xn2){
  int tid = threadIdx.x; int wv = tid >> 5, lane = tid & 31;
  size_t t = (size_t)blockIdx.x * 8 + wv;
  float v[4]; float s = 0.f;
#pragma unroll
  for (int j = 0; j < 4; ++j){
    int c = lane + 32*j;
    v[j] = xh[t*128 + c] + ls1[c] * pout[t*128 + c];
    s += v[j];
  }
  float mu = wred_sum(s) * (1.f/128.f);
  float q = 0.f;
#pragma unroll
  for (int j = 0; j < 4; ++j){ float d = v[j]-mu; q += d*d; }
  float r = rsqrtf(wred_sum(q) * (1.f/128.f) + 1e-6f);
#pragma unroll
  for (int j = 0; j < 4; ++j){
    int c = lane + 32*j;
    xh[t*128 + c]  = v[j];
    xn2[t*128 + c] = (bf16)(((v[j]-mu)*r) * g[c] + bta[c]);
  }
}

// ---------------- mid LN (over 170) + GELU -> bf16 (padded 192) -------------------
__global__ __launch_bounds__(256) void k_mid(const float* __restrict__ hsrc,
        const float* __restrict__ mg, const float* __restrict__ mb,
        bf16* __restrict__ hb){
  int tid = threadIdx.x; int wv = tid >> 5, lane = tid & 31;
  size_t t = (size_t)blockIdx.x * 8 + wv;
  const float* row = hsrc + t * HIDP;
  float v[6]; float s = 0.f;
#pragma unroll
  for (int j = 0; j < 6; ++j){
    int c = lane + 32*j;
    v[j] = (c < HIDR) ? row[c] : 0.f;
    s += v[j];
  }
  float mu = wred_sum(s) * (1.f/(float)HIDR);
  float q = 0.f;
#pragma unroll
  for (int j = 0; j < 6; ++j){
    int c = lane + 32*j;
    if (c < HIDR){ float d = v[j]-mu; q += d*d; }
  }
  float r = rsqrtf(wred_sum(q) * (1.f/(float)HIDR) + 1e-6f);
#pragma unroll
  for (int j = 0; j < 6; ++j){
    int c = lane + 32*j;
    float o = 0.f;
    if (c < HIDR){
      float xc = (v[j]-mu)*r * mg[c] + mb[c];
      o = 0.5f*xc*(1.f + tanhf(0.7978845608028654f*(xc + 0.044715f*xc*xc*xc)));
    }
    hb[t*HIDP + c] = (bf16)o;
  }
}

// ---------------- residual 2 + transpose out --------------------------------------
__global__ __launch_bounds__(256) void k_resid2(const float* __restrict__ xh,
        const float* __restrict__ fo, const float* __restrict__ ls2,
        float* __restrict__ out){
  __shared__ float tile[128][65];
  int tid = threadIdx.x;
  int t0  = blockIdx.x * 64;
  int b   = t0 / HWSZ;
  int hw0 = t0 % HWSZ;
  for (int e = tid; e < 64*128; e += 256){
    int t = e >> 7, c = e & 127;
    size_t idx = (size_t)(t0 + t)*128 + c;
    tile[c][t] = xh[idx] + ls2[c]*fo[idx];
  }
  __syncthreads();
  float* ob = out + (size_t)b * CCH * HWSZ + hw0;
  for (int e = tid; e < 128*64; e += 256){
    int c = e >> 6, t = e & 63;
    ob[(size_t)c * HWSZ + t] = tile[c][t];
  }
}

// ---------------- weight conversion helpers ---------------------------------------
__global__ void k_cvt(const float* __restrict__ s, bf16* __restrict__ d, int n){
  int i = blockIdx.x*256 + threadIdx.x;
  if (i < n) d[i] = (bf16)s[i];
}
__global__ void k_cvt_fc1w(const float* __restrict__ s, bf16* __restrict__ d){
  int i = blockIdx.x*256 + threadIdx.x;
  if (i < HIDP*128){
    int rr = i >> 7, c = i & 127;
    d[i] = (bf16)((rr < HIDR) ? s[rr*128 + c] : 0.f);
  }
}
__global__ void k_cvt_fc2w(const float* __restrict__ s, bf16* __restrict__ d){
  int i = blockIdx.x*256 + threadIdx.x;
  if (i < 128*HIDP){
    int rr = i / HIDP, c = i % HIDP;
    d[i] = (bf16)((c < HIDR) ? s[rr*HIDR + c] : 0.f);
  }
}
__global__ void k_pad_fc1b(const float* __restrict__ s, float* __restrict__ d){
  int i = threadIdx.x;
  if (i < HIDP) d[i] = (i < HIDR) ? s[i] : 0.f;
}

// =================================================================================
extern "C" void kernel_launch(void* const* d_in, const int* in_sizes, int n_in,
                              void* d_out, int out_size, void* d_ws, size_t ws_size,
                              hipStream_t stream){
  (void)in_sizes; (void)n_in; (void)out_size; (void)ws_size;
  const float* x      = (const float*)d_in[0];
  const float* qkv_w  = (const float*)d_in[1];
  const float* qkv_b  = (const float*)d_in[2];
  const float* qg     = (const float*)d_in[3];
  const float* qb     = (const float*)d_in[4];
  const float* kg     = (const float*)d_in[5];
  const float* kb     = (const float*)d_in[6];
  const float* proj_w = (const float*)d_in[7];
  const float* proj_b = (const float*)d_in[8];
  const float* n1g    = (const float*)d_in[9];
  const float* n1b    = (const float*)d_in[10];
  const float* n2g    = (const float*)d_in[11];
  const float* n2b    = (const float*)d_in[12];
  const float* ls1    = (const float*)d_in[13];
  const float* ls2    = (const float*)d_in[14];
  const float* fc1_w  = (const float*)d_in[15];
  const float* fc1_b  = (const float*)d_in[16];
  const float* mg     = (const float*)d_in[17];
  const float* mb     = (const float*)d_in[18];
  const float* fc2_w  = (const float*)d_in[19];
  const float* fc2_b  = (const float*)d_in[20];

  char* ws = (char*)d_ws;
  size_t off = 0;
  auto take = [&](size_t bytes)->size_t{
    size_t r = off; off = (off + bytes + 255) & ~(size_t)255; return r;
  };
  const size_t M = MTOK;
  size_t o_xh  = take(M*128*4);
  size_t o_xn  = take(M*128*2);
  size_t o_big = take(M*384*4);                  // qkv; later pout|hbuf; later hb
  size_t o_qb  = take(M*128*2);                  // q bf16; later fout (spans next bufs)
  size_t o_kb  = take((size_t)BB*HC*WC*128*2);
  size_t o_vb  = take((size_t)BB*HC*WC*128*2);
  size_t o_ao  = take(M*128*2);
  size_t o_wq  = take(384*128*2);
  size_t o_wp  = take(128*128*2);
  size_t o_w1  = take((size_t)HIDP*128*2);
  size_t o_w2  = take((size_t)128*HIDP*2);
  size_t o_b1  = take(HIDP*4);

  float* xh   = (float*)(ws + o_xh);
  bf16*  xn   = (bf16*)(ws + o_xn);
  bf16*  xn2  = (bf16*)(ws + o_xn);              // reuse (xn dead after QKV gemm)
  float* qkv  = (float*)(ws + o_big);
  float* pout = (float*)(ws + o_big);            // reuse (qkv dead after qkprep)
  float* hbuf = (float*)(ws + o_big + M*128*4);
  bf16*  hb   = (bf16*)(ws + o_big);             // reuse (pout dead after resid1)
  bf16*  qbm  = (bf16*)(ws + o_qb);
  bf16*  kbm  = (bf16*)(ws + o_kb);
  bf16*  vbm  = (bf16*)(ws + o_vb);
  bf16*  aom  = (bf16*)(ws + o_ao);
  float* fout = (float*)(ws + o_qb);             // reuse (q/k/v/aout dead by fc2)
  bf16*  wq   = (bf16*)(ws + o_wq);
  bf16*  wp   = (bf16*)(ws + o_wp);
  bf16*  w1   = (bf16*)(ws + o_w1);
  bf16*  w2   = (bf16*)(ws + o_w2);
  float* b1p  = (float*)(ws + o_b1);

  // weight prep
  k_cvt     <<<(384*128+255)/256, 256, 0, stream>>>(qkv_w, wq, 384*128);
  k_cvt     <<<(128*128+255)/256, 256, 0, stream>>>(proj_w, wp, 128*128);
  k_cvt_fc1w<<<(HIDP*128+255)/256, 256, 0, stream>>>(fc1_w, w1);
  k_cvt_fc2w<<<(128*HIDP+255)/256, 256, 0, stream>>>(fc2_w, w2);
  k_pad_fc1b<<<1, 256, 0, stream>>>(fc1_b, b1p);

  // pipeline
  k_ln1   <<<M/64, 256, 0, stream>>>(x, n1g, n1b, xh, xn);
  k_gemm  <<<dim3(M/256, 384/64), 256, 0, stream>>>(xn,  wq, qkv_b, qkv,  128, 384);
  k_qkprep<<<M/8, 256, 0, stream>>>(qkv, qg, qb, kg, kb, qbm, kbm, vbm);
  k_attn  <<<BB*HC*WC, 256, 0, stream>>>(qbm, kbm, vbm, aom);
  k_gemm  <<<dim3(M/256, 128/64), 256, 0, stream>>>(aom, wp, proj_b, pout, 128, 128);
  k_resid1<<<M/8, 256, 0, stream>>>(pout, ls1, n2g, n2b, xh, xn2);
  k_gemm  <<<dim3(M/256, HIDP/64), 256, 0, stream>>>(xn2, w1, b1p, hbuf, 128, HIDP);
  k_mid   <<<M/8, 256, 0, stream>>>(hbuf, mg, mb, hb);
  k_gemm  <<<dim3(M/256, 128/64), 256, 0, stream>>>(hb, w2, fc2_b, fout, HIDP, 128);
  k_resid2<<<M/64, 256, 0, stream>>>(xh, fout, ls2, (float*)d_out);
}
